// MultiHeadMoeBlock_44667659878793
// MI455X (gfx1250) — matrix operand order repaired and checked by reading, not compiled, not run on hardware
//
#include <hip/hip_runtime.h>
#include <hip/hip_bf16.h>
#include <math.h>

// ---------------- problem constants ----------------
#define T_TOK  65536   // BS*L*H sub-tokens
#define HDIM   128     // per-head dim (HD)
#define DMODEL 1024    // D
#define MROWS  8192    // BS*L
#define NEXP   16      // E
#define IDIM   512     // per-expert intermediate I
#define TOPK   2
#define MTILE  32      // tokens per MoE-MLP tile

// ---------------- vector types ----------------
typedef __bf16 v16bf __attribute__((ext_vector_type(16)));
typedef float  v8f   __attribute__((ext_vector_type(8)));
typedef unsigned short u16x8 __attribute__((ext_vector_type(8)));
typedef unsigned short u16x4 __attribute__((ext_vector_type(4)));
typedef float          f32x4 __attribute__((ext_vector_type(4)));
typedef int            v4i   __attribute__((vector_size(16)));   // matches builtin param

#if __has_builtin(__builtin_amdgcn_global_load_async_to_lds_b128)
#define USE_ASYNC_LDS 1
typedef __attribute__((address_space(1))) v4i* as1_v4i;
typedef __attribute__((address_space(3))) v4i* as3_v4i;
#else
#define USE_ASYNC_LDS 0
#endif

static __device__ __forceinline__ void wait_asynccnt(int n) {
#if __has_builtin(__builtin_amdgcn_s_wait_asynccnt)
  if (n == 0) __builtin_amdgcn_s_wait_asynccnt(0);
  else        __builtin_amdgcn_s_wait_asynccnt(2);
#else
  asm volatile("s_wait_asynccnt 0" ::: "memory");
#endif
}

static __device__ __forceinline__ unsigned short f2bf(float f) {
  unsigned int u = __float_as_uint(f);
  unsigned int r = (u + 0x7FFFu + ((u >> 16) & 1u)) >> 16;   // round-nearest-even
  return (unsigned short)r;
}
static __device__ __forceinline__ float bf2f(unsigned short h) {
  return __uint_as_float(((unsigned int)h) << 16);
}

union FragBF { v16bf v; u16x8 h[2]; };

// A-matrix fragment (16x32 bf16): lane L holds row L%16, half=L/16.
// element j -> K = 8*half + 16*(j>>3) + (j&7)  => two contiguous 8-elem chunks.
static __device__ __forceinline__ v16bf load_a_frag(const unsigned short* A, int ldk,
                                                    int row, int k0, int half) {
  FragBF f;
  const unsigned short* p = A + row * ldk + k0 + 8 * half;
  f.h[0] = *(const u16x8*)p;
  f.h[1] = *(const u16x8*)(p + 16);
  return f.v;
}
// B-matrix fragment (32x16 bf16) from N-major weight [n][k]:
// lane L holds column n=L%16; element j -> K = 8*half + 16*(j>>3) + (j&7).
static __device__ __forceinline__ v16bf load_b_frag(const unsigned short* Bt, int K,
                                                    int n, int k0, int half) {
  FragBF f;
  const unsigned short* p = Bt + (size_t)(n + (threadIdx.x & 15)) * K + k0 + 8 * half;
  f.h[0] = *(const u16x8*)p;
  f.h[1] = *(const u16x8*)(p + 16);
  return f.v;
}
static __device__ __forceinline__ v8f wmma_bf16(v16bf a, v16bf b, v8f c) {
  return __builtin_amdgcn_wmma_f32_16x16x32_bf16(false, a, false, b, (short)0, c,
                                                 false, false);
}

// ---------------- elementwise f32 -> bf16 ----------------
__global__ __launch_bounds__(256) void conv_kernel(const float* __restrict__ src,
                                                   unsigned short* __restrict__ dst,
                                                   int n) {
  int i = blockIdx.x * 256 + threadIdx.x;
  if (i < n) dst[i] = f2bf(src[i]);
}

// ---------------- weight convert + transpose to bf16 N-major ----------------
// in:  W  [E][K][N] f32   out: Wt [E][N][K] bf16
__global__ __launch_bounds__(256) void convt_kernel(const float* __restrict__ W,
                                                    unsigned short* __restrict__ Wt,
                                                    int E, int K, int N) {
  long long idx = (long long)blockIdx.x * 256 + threadIdx.x;
  long long total = (long long)E * K * N;
  if (idx >= total) return;
  int kn = K * N;
  int e = (int)(idx / kn);
  int r = (int)(idx % kn);
  int n = r / K, k = r % K;
  Wt[idx] = f2bf(W[((size_t)e * K + k) * N + n]);
}

// ---------------- big GEMM: C[M,N] = A[M,K](bf16) * Bt[N,K](bf16) + bias -----
// block = 128 threads (4 waves). Tile: 64(M) x 256(N); wave covers 64 N cols,
// all 4 M sub-tiles. A tile double-buffered via async global->LDS.
template <bool OUT_BF16>
__global__ __launch_bounds__(128) void gemm_bf16_kernel(
    const unsigned short* __restrict__ A, const unsigned short* __restrict__ Bt,
    const float* __restrict__ bias, void* __restrict__ Out, int M, int N, int K) {
  __shared__ unsigned short Abuf[2][64 * 32];   // 2 x 4 KB
  const int tid = threadIdx.x;
  const int wave = tid >> 5, lane = tid & 31;
  const int m0 = blockIdx.y * 64;
  const int nbase = blockIdx.x * 256 + wave * 64;

  const int sr = tid >> 1, sc = (tid & 1) * 16;   // staging row/col for this thread

  auto stage = [&](int buf, int k0) {
    const unsigned short* g = A + (size_t)(m0 + sr) * K + k0 + sc;
    unsigned short* l = &Abuf[buf][sr * 32 + sc];
#if USE_ASYNC_LDS
    __builtin_amdgcn_global_load_async_to_lds_b128(
        (as1_v4i)(uintptr_t)g, (as3_v4i)(unsigned)(uintptr_t)l, 0, 0);
    __builtin_amdgcn_global_load_async_to_lds_b128(
        (as1_v4i)(uintptr_t)(g + 8), (as3_v4i)(unsigned)(uintptr_t)(l + 8), 0, 0);
#else
    const u16x8* s = (const u16x8*)g;
    u16x8* d = (u16x8*)l;
    d[0] = s[0];
    d[1] = s[1];
#endif
  };

  v8f vzero = {0.f, 0.f, 0.f, 0.f, 0.f, 0.f, 0.f, 0.f};
  v8f acc[4][4];
#pragma unroll
  for (int mt = 0; mt < 4; ++mt)
#pragma unroll
    for (int nt = 0; nt < 4; ++nt) acc[mt][nt] = vzero;

  stage(0, 0);  // prologue: first A tile in flight
  int it = 0;
  for (int k0 = 0; k0 < K; k0 += 32, ++it) {
    const int cur = it & 1;
    __syncthreads();  // prior reads of buf[cur^1] done -> safe to overwrite it
    if (k0 + 32 < K) {
      stage(cur ^ 1, k0 + 32);   // prefetch next tile (overlaps with compute)
      wait_asynccnt(2);          // in-order: <=2 outstanding => buf[cur] complete
    } else {
      wait_asynccnt(0);
    }
    __syncthreads();  // all waves' copies for buf[cur] visible

    v16bf a[4];
#pragma unroll
    for (int mt = 0; mt < 4; ++mt)
      a[mt] = load_a_frag(Abuf[cur], 32, mt * 16 + (lane & 15), 0, lane >> 4);
#pragma unroll
    for (int nt = 0; nt < 4; ++nt) {
      v16bf b = load_b_frag(Bt, K, nbase + nt * 16, k0, lane >> 4);
#pragma unroll
      for (int mt = 0; mt < 4; ++mt) acc[mt][nt] = wmma_bf16(a[mt], b, acc[mt][nt]);
    }
  }
  // epilogue: C layout -> lane L holds rows v+8*(L/16), col L%16
#pragma unroll
  for (int mt = 0; mt < 4; ++mt) {
#pragma unroll
    for (int nt = 0; nt < 4; ++nt) {
      int n = nbase + nt * 16 + (lane & 15);
      float bv = bias[n];
#pragma unroll
      for (int v = 0; v < 8; ++v) {
        int m = m0 + mt * 16 + v + 8 * (lane >> 4);
        float val = acc[mt][nt][v] + bv;
        if (OUT_BF16)
          ((unsigned short*)Out)[(size_t)m * N + n] = f2bf(val);
        else
          ((float*)Out)[(size_t)m * N + n] = val;
      }
    }
  }
}

// ---------------- routing: one wave per sub-token ----------------
__global__ __launch_bounds__(256) void route_kernel(
    const unsigned short* __restrict__ hbf, const float* __restrict__ emb,
    int* __restrict__ topi, float* __restrict__ topw, int* __restrict__ counts) {
  const int wave = threadIdx.x >> 5, lane = threadIdx.x & 31;
  const int t = blockIdx.x * 8 + wave;

  float tv[4];
#pragma unroll
  for (int j = 0; j < 4; ++j) tv[j] = bf2f(hbf[(size_t)t * HDIM + lane * 4 + j]);

  float logits[16];
#pragma unroll
  for (int e = 0; e < 16; ++e) {
    float p = 0.f;
#pragma unroll
    for (int j = 0; j < 4; ++j) p += tv[j] * emb[(lane * 4 + j) * NEXP + e];
    p += __shfl_xor(p, 16, 32);
    p += __shfl_xor(p, 8, 32);
    p += __shfl_xor(p, 4, 32);
    p += __shfl_xor(p, 2, 32);
    p += __shfl_xor(p, 1, 32);
    logits[e] = p;
  }
  if (lane == 0) {
    float mx = logits[0];
#pragma unroll
    for (int e = 1; e < 16; ++e) mx = fmaxf(mx, logits[e]);
    float pr[16], s = 0.f;
#pragma unroll
    for (int e = 0; e < 16; ++e) { pr[e] = expf(logits[e] - mx); s += pr[e]; }
    float inv = 1.f / s;
    int i0 = 0; float w0 = pr[0];
#pragma unroll
    for (int e = 1; e < 16; ++e) if (pr[e] > w0) { w0 = pr[e]; i0 = e; }
    int i1 = (i0 == 0) ? 1 : 0; float w1 = pr[i1];
#pragma unroll
    for (int e = 0; e < 16; ++e)
      if (e != i0 && pr[e] > w1) { w1 = pr[e]; i1 = e; }
    topi[2 * t] = i0; topi[2 * t + 1] = i1;
    topw[2 * t] = w0 * inv; topw[2 * t + 1] = w1 * inv;
    atomicAdd(&counts[i0], 1);
    atomicAdd(&counts[i1], 1);
  }
}

// ---------------- tiny scan over 16 experts ----------------
__global__ void scan_kernel(const int* __restrict__ counts, int* __restrict__ offsets,
                            int* __restrict__ cursor, int* __restrict__ tileOff) {
  if (threadIdx.x == 0 && blockIdx.x == 0) {
    int off = 0, toff = 0;
    for (int e = 0; e < NEXP; ++e) {
      offsets[e] = off; cursor[e] = off; tileOff[e] = toff;
      off += counts[e];
      toff += (counts[e] + (MTILE - 1)) / MTILE;
    }
    tileOff[NEXP] = toff;
  }
}

// ---------------- scatter (token,k) -> expert buckets ----------------
__global__ __launch_bounds__(256) void scatter_kernel(
    const int* __restrict__ topi, const float* __restrict__ topw,
    int* __restrict__ cursor, int* __restrict__ btok, float* __restrict__ bw) {
  int t = blockIdx.x * 256 + threadIdx.x;
  if (t >= T_TOK) return;
#pragma unroll
  for (int k = 0; k < TOPK; ++k) {
    int e = topi[2 * t + k];
    int pos = atomicAdd(&cursor[e], 1);
    btok[pos] = t;
    bw[pos] = topw[2 * t + k];
  }
}

// ---------------- per-expert MLP on 32-token tiles (all WMMA) ----------------
__global__ __launch_bounds__(128) void moe_mlp_kernel(
    const unsigned short* __restrict__ hbf,
    const unsigned short* __restrict__ W1t, const float* __restrict__ b1,
    const unsigned short* __restrict__ W2t, const float* __restrict__ b2,
    const int* __restrict__ btok, const float* __restrict__ bw,
    const int* __restrict__ offsets, const int* __restrict__ counts,
    const int* __restrict__ tileOff, float* __restrict__ comb) {
  __shared__ unsigned short Atile[MTILE * HDIM];   // 8 KB
  __shared__ unsigned short Mid[MTILE * IDIM];     // 32 KB
  __shared__ int sTok[MTILE];
  __shared__ float sW[MTILE];

  const int b = blockIdx.x;
  int e = -1;
#pragma unroll
  for (int i = 0; i < NEXP; ++i)
    if (b >= tileOff[i] && b < tileOff[i + 1]) e = i;
  if (e < 0) return;

  const int tile_in = b - tileOff[e];
  const int cnt = counts[e];
  const int base = offsets[e] + tile_in * MTILE;
  int nvalid = cnt - tile_in * MTILE;
  if (nvalid > MTILE) nvalid = MTILE;

  const int tid = threadIdx.x;
  const int wave = tid >> 5, lane = tid & 31;
  const int half = lane >> 4, lan16 = lane & 15;

  if (tid < MTILE) {
    if (tid < nvalid) { sTok[tid] = btok[base + tid]; sW[tid] = bw[base + tid]; }
    else { sTok[tid] = 0; sW[tid] = 0.f; }
  }
  {  // gather 32 token rows (bf16) into LDS: thread -> row tid/4, 32-elem chunk
    int r = tid >> 2, c = (tid & 3) * 32;
    u16x8* d = (u16x8*)(Atile + r * HDIM + c);
    if (r < nvalid) {
      int tok = btok[base + r];
      const u16x8* s = (const u16x8*)(hbf + (size_t)tok * HDIM + c);
      d[0] = s[0]; d[1] = s[1]; d[2] = s[2]; d[3] = s[3];
    } else {
      u16x8 z = {0, 0, 0, 0, 0, 0, 0, 0};
      d[0] = z; d[1] = z; d[2] = z; d[3] = z;
    }
  }
  __syncthreads();

  v8f vzero = {0.f, 0.f, 0.f, 0.f, 0.f, 0.f, 0.f, 0.f};
  // ---- stage 1: Mid = gelu(A(32x128) @ W1[e](128x512) + b1[e]) ----
  const unsigned short* W1e = W1t + (size_t)e * IDIM * HDIM;  // [n=512][k=128]
  const float* b1e = b1 + e * IDIM;
  v8f acc1[2][8];
#pragma unroll
  for (int mt = 0; mt < 2; ++mt)
#pragma unroll
    for (int i = 0; i < 8; ++i) acc1[mt][i] = vzero;
  for (int k0 = 0; k0 < HDIM; k0 += 32) {
    v16bf a0 = load_a_frag(Atile, HDIM, lan16, k0, half);
    v16bf a1 = load_a_frag(Atile, HDIM, 16 + lan16, k0, half);
#pragma unroll
    for (int nt = 0; nt < 8; ++nt) {
      v16bf bb = load_b_frag(W1e, HDIM, wave * 128 + nt * 16, k0, half);
      acc1[0][nt] = wmma_bf16(a0, bb, acc1[0][nt]);
      acc1[1][nt] = wmma_bf16(a1, bb, acc1[1][nt]);
    }
  }
#pragma unroll
  for (int mt = 0; mt < 2; ++mt) {
#pragma unroll
    for (int nt = 0; nt < 8; ++nt) {
      int n = wave * 128 + nt * 16 + lan16;
      float bv = b1e[n];
#pragma unroll
      for (int v = 0; v < 8; ++v) {
        int m = mt * 16 + v + 8 * half;
        float x = acc1[mt][nt][v] + bv;
        float g = 0.5f * x * (1.0f + erff(x * 0.70710678118654752f));  // exact gelu
        Mid[m * IDIM + n] = f2bf(g);
      }
    }
  }
  __syncthreads();

  // ---- stage 2: out = Mid(32x512) @ W2[e](512x128) + b2[e] ----
  const unsigned short* W2e = W2t + (size_t)e * HDIM * IDIM;  // [n=128][k=512]
  const float* b2e = b2 + e * HDIM;
  v8f acc2[2][2];
  acc2[0][0] = vzero; acc2[0][1] = vzero; acc2[1][0] = vzero; acc2[1][1] = vzero;
  for (int k0 = 0; k0 < IDIM; k0 += 32) {
    v16bf a0 = load_a_frag(Mid, IDIM, lan16, k0, half);
    v16bf a1 = load_a_frag(Mid, IDIM, 16 + lan16, k0, half);
#pragma unroll
    for (int nt = 0; nt < 2; ++nt) {
      v16bf bb = load_b_frag(W2e, IDIM, wave * 32 + nt * 16, k0, half);
      acc2[0][nt] = wmma_bf16(a0, bb, acc2[0][nt]);
      acc2[1][nt] = wmma_bf16(a1, bb, acc2[1][nt]);
    }
  }
#pragma unroll
  for (int mt = 0; mt < 2; ++mt) {
#pragma unroll
    for (int nt = 0; nt < 2; ++nt) {
      int n = wave * 32 + nt * 16 + lan16;
      float bv = b2e[n];
#pragma unroll
      for (int v = 0; v < 8; ++v) {
        int m = mt * 16 + v + 8 * half;
        if (m < nvalid) {
          float val = (acc2[mt][nt][v] + bv) * sW[m];
          atomicAdd(&comb[(size_t)sTok[m] * HDIM + n], val);
        }
      }
    }
  }
}

// ---------------- launch ----------------
extern "C" void kernel_launch(void* const* d_in, const int* in_sizes, int n_in,
                              void* d_out, int out_size, void* d_ws, size_t ws_size,
                              hipStream_t stream) {
  (void)in_sizes; (void)n_in; (void)out_size; (void)ws_size;
  const float* x   = (const float*)d_in[0];
  const float* Wm  = (const float*)d_in[1];
  const float* bm  = (const float*)d_in[2];
  const float* emb = (const float*)d_in[3];
  const float* W1  = (const float*)d_in[4];
  const float* b1  = (const float*)d_in[5];
  const float* W2  = (const float*)d_in[6];
  const float* b2  = (const float*)d_in[7];
  const float* Wo  = (const float*)d_in[8];
  const float* bo  = (const float*)d_in[9];
  float* out = (float*)d_out;

  char* ws = (char*)d_ws;
  size_t off = 0;
  auto alloc = [&](size_t bytes) -> void* {
    void* p = ws + off;
    off += bytes;
    off = (off + 255) & ~(size_t)255;
    return p;
  };
  unsigned short* hbf = (unsigned short*)alloc((size_t)T_TOK * HDIM * 2);
  unsigned short* abf = (unsigned short*)alloc((size_t)MROWS * DMODEL * 2);  // x_bf16, then comb_bf16
  unsigned short* WmT = (unsigned short*)alloc((size_t)DMODEL * DMODEL * 2);
  unsigned short* WoT = (unsigned short*)alloc((size_t)DMODEL * DMODEL * 2);
  unsigned short* W1T = (unsigned short*)alloc((size_t)NEXP * IDIM * HDIM * 2);
  unsigned short* W2T = (unsigned short*)alloc((size_t)NEXP * HDIM * IDIM * 2);
  int*   topi   = (int*)alloc((size_t)T_TOK * 2 * 4);
  float* topw   = (float*)alloc((size_t)T_TOK * 2 * 4);
  int*   btok   = (int*)alloc((size_t)T_TOK * 2 * 4);
  float* bwgt   = (float*)alloc((size_t)T_TOK * 2 * 4);
  int* counts   = (int*)alloc(64);
  int* offsets  = (int*)alloc(64);
  int* cursor   = (int*)alloc(64);
  int* tileOff  = (int*)alloc(128);
  float* comb   = (float*)alloc((size_t)T_TOK * HDIM * 4);

  (void)hipMemsetAsync(counts, 0, 64, stream);
  (void)hipMemsetAsync(comb, 0, (size_t)T_TOK * HDIM * 4, stream);

  // weights -> bf16, transposed N-major; x -> bf16
  convt_kernel<<<(DMODEL * DMODEL + 255) / 256, 256, 0, stream>>>(Wm, WmT, 1, DMODEL, DMODEL);
  convt_kernel<<<(DMODEL * DMODEL + 255) / 256, 256, 0, stream>>>(Wo, WoT, 1, DMODEL, DMODEL);
  convt_kernel<<<(NEXP * HDIM * IDIM + 255) / 256, 256, 0, stream>>>(W1, W1T, NEXP, HDIM, IDIM);
  convt_kernel<<<(NEXP * IDIM * HDIM + 255) / 256, 256, 0, stream>>>(W2, W2T, NEXP, IDIM, HDIM);
  conv_kernel<<<(MROWS * DMODEL + 255) / 256, 256, 0, stream>>>(x, abf, MROWS * DMODEL);

  dim3 ggrid(DMODEL / 256, MROWS / 64);
  // h = x @ Wm + bm   (bf16 out -> hbf, which is also the [T, HD] sub-token view)
  gemm_bf16_kernel<true><<<ggrid, 128, 0, stream>>>(abf, WmT, bm, hbf, MROWS, DMODEL, DMODEL);

  route_kernel<<<T_TOK / 8, 256, 0, stream>>>(hbf, emb, topi, topw, counts);
  scan_kernel<<<1, 32, 0, stream>>>(counts, offsets, cursor, tileOff);
  scatter_kernel<<<T_TOK / 256, 256, 0, stream>>>(topi, topw, cursor, btok, bwgt);

  int maxTiles = (T_TOK * TOPK) / MTILE + NEXP;  // upper bound on sum ceil(cnt/MTILE)
  moe_mlp_kernel<<<maxTiles, 128, 0, stream>>>(hbf, W1T, b1, W2T, b2, btok, bwgt,
                                               offsets, counts, tileOff, comb);

  // out = comb(view [8192,1024]) @ Wo + bo  (f32 out)
  conv_kernel<<<(MROWS * DMODEL + 255) / 256, 256, 0, stream>>>(comb, abf, MROWS * DMODEL);
  gemm_bf16_kernel<false><<<ggrid, 128, 0, stream>>>(abf, WoT, bo, out, MROWS, DMODEL, DMODEL);
}
